// NonRecurrentLSTM_47614007443935
// MI455X (gfx1250) — matrix operand order repaired and checked by reading, not compiled, not run on hardware
//
#include <hip/hip_runtime.h>
#include <stdint.h>

// ---------------------------------------------------------------------------
// NonRecurrentLSTM for MI455X (gfx1250, wave32, WMMA).
//
// gates = X @ W^T + b ; i=sig, g=tanh, o=sig ; c=i*g ; h=o*tanh(c)
// f-gate is dead in the reference -> its quarter of the GEMM is skipped.
//
// Fused per layer/direction: bf16 WMMA GEMM (fp32 acc) for gates {i,g,o}
// + activation epilogue. Wave tile 64x32 per gate (24 WMMA / 10KB fragment
// loads = 38 FLOP/B), explicit double-buffered fragment sets, and
// SGPR-base + loop-invariant 32-bit BYTE-offset VGPR addressing (GVS form)
// so the K-loop has no VALU address math and no WMMA->VALU WAR v_nops.
// ---------------------------------------------------------------------------

typedef __attribute__((ext_vector_type(16))) __bf16 v16bf;
typedef __attribute__((ext_vector_type(8)))  float  v8f;

union FragU { uint4 u[2]; v16bf v; };

__device__ __forceinline__ unsigned short f2bf(float f) {
    unsigned int u = __float_as_uint(f);
    u += 0x7FFFu + ((u >> 16) & 1u);        // round-to-nearest-even
    return (unsigned short)(u >> 16);
}

__device__ __forceinline__ float sigmoidf_(float x) {
    return 1.0f / (1.0f + __expf(-x));
}

// A fragment, 16x32 bf16 (ISA 7.12.2): lane L -> row = L%16;
// half-wave 0 holds K-chunks [0,8) and [16,24); half-wave 1: [8,16),[24,32).
// base_k = X + k0 bytes (uniform, SGPR); boff = byte offset (divergent,
// loop-invariant 32-bit VGPR) -> global_load saddr+voffset (GVS) form.
__device__ __forceinline__ v16bf load_fragA3(const char* __restrict__ base_k,
                                             uint32_t boff) {
    FragU f;
    f.u[0] = *(const uint4*)(base_k + boff);        // K chunk c
    f.u[1] = *(const uint4*)(base_k + boff + 32);   // K chunk c+16 (bf16)
    return f.v;
}

// B fragment, 32x16 bf16 (K x N): same K split as A (half-wave 0: [0,8),[16,24);
// half-wave 1: [8,16),[24,32)) of column n = L%16. B[k,n] = W[n,k].
__device__ __forceinline__ v16bf load_fragB3(const char* __restrict__ base_k,
                                             uint32_t boff) {
    FragU f;
    f.u[0] = *(const uint4*)(base_k + boff);
    f.u[1] = *(const uint4*)(base_k + boff + 32);
    return f.v;
}

__global__ void cvt_f32_bf16_v4(const float4* __restrict__ in,
                                ushort4* __restrict__ out, int n4) {
    int i = blockIdx.x * blockDim.x + threadIdx.x;
    int stride = gridDim.x * blockDim.x;
    for (; i < n4; i += stride) {
        float4 v = in[i];
        ushort4 o;
        o.x = f2bf(v.x); o.y = f2bf(v.y); o.z = f2bf(v.z); o.w = f2bf(v.w);
        out[i] = o;
    }
}

// Block: 256 threads = 8 waves arranged 4 (M) x 2 (N).
// Block tile 256 (M) x 64 (N per gate) x 3 gates; wave tile 64x32.
// acc = 192 VGPRs + double-buffered fragments = 160 VGPRs (~380 total;
// gfx1250's 1024-VGPR file via VGPR-MSB -> 2 waves/SIMD).
template<int K, bool OUT_BF16>
__global__ __launch_bounds__(256)
void lstm_layer_kernel(const uint16_t* __restrict__ X,    // [16384, K] bf16
                       const uint16_t* __restrict__ Wf,   // [4096, K] bf16
                       const uint16_t* __restrict__ Wb,   // [4096, K] bf16
                       const float* __restrict__ biasf,   // [4096] f32
                       const float* __restrict__ biasb,   // [4096] f32
                       void* __restrict__ outp,           // [16384, 2048]
                       float* __restrict__ hn,            // [4, 8, 1024]
                       float* __restrict__ cn,            // [4, 8, 1024]
                       int layer) {
    const int lane = threadIdx.x & 31;
    const int wid  = threadIdx.x >> 5;
    const int ln15 = lane & 15;
    const int hl   = lane >> 4;

    const int row0 = blockIdx.x * 256;   // M tile
    const int j0   = blockIdx.y * 64;    // hidden-column tile (within H=1024)
    const int dir  = blockIdx.z;         // 0 = forward, 1 = backward

    const uint16_t* W = dir ? Wb : Wf;
    const float* bias = dir ? biasb : biasf;

    const int mw = (wid >> 1) * 64;      // wave M offset within block tile
    const int nw = (wid & 1) * 32;       // wave N offset within block tile

    const int grow[3] = {0, 2048, 3072}; // weight-row offsets of gates i, g, o

    // Loop-invariant divergent 32-bit BYTE offsets for each fragment.
    uint32_t aoff[4];
    #pragma unroll
    for (int mf = 0; mf < 4; ++mf)
        aoff[mf] = (uint32_t)(((row0 + mw + mf * 16 + ln15) * K + hl * 8) * 2);
    uint32_t boff[6];
    #pragma unroll
    for (int nf = 0; nf < 2; ++nf)
        #pragma unroll
        for (int g = 0; g < 3; ++g)
            boff[nf * 3 + g] = (uint32_t)(
                ((grow[g] + j0 + nw + nf * 16 + ln15) * K + hl * 8) * 2);

    // Init accumulators with bias (each lane's column n is constant per acc).
    v8f acc[4][2][3];
    #pragma unroll
    for (int nf = 0; nf < 2; ++nf) {
        #pragma unroll
        for (int g = 0; g < 3; ++g) {
            float bv = bias[grow[g] + j0 + nw + nf * 16 + ln15];
            v8f a;
            #pragma unroll
            for (int r = 0; r < 8; ++r) a[r] = bv;
            #pragma unroll
            for (int mf = 0; mf < 4; ++mf) acc[mf][nf][g] = a;
        }
    }

    // Double-buffered fragment sets (ping-pong): loads for set 1-s land in
    // registers disjoint from the set the WMMA burst is consuming.
    v16bf aF[2][4];
    v16bf bF[2][6];

    auto load_set = [&](int s, int k0) {
        const char* xk = (const char*)X + (size_t)k0 * 2;  // uniform -> SGPR
        const char* wk = (const char*)W + (size_t)k0 * 2;
        #pragma unroll
        for (int mf = 0; mf < 4; ++mf) aF[s][mf] = load_fragA3(xk, aoff[mf]);
        #pragma unroll
        for (int q = 0; q < 6; ++q)    bF[s][q]  = load_fragB3(wk, boff[q]);
    };
    auto compute_set = [&](int s) {
        #pragma unroll
        for (int nf = 0; nf < 2; ++nf) {
            #pragma unroll
            for (int g = 0; g < 3; ++g) {
                #pragma unroll
                for (int mf = 0; mf < 4; ++mf) {
                    acc[mf][nf][g] = __builtin_amdgcn_wmma_f32_16x16x32_bf16(
                        false, aF[s][mf], false, bF[s][nf * 3 + g], (short)0,
                        acc[mf][nf][g], false, false);
                }
            }
        }
    };

    // K is a multiple of 64 (1024 or 2048): 2-stage pipeline, even chunk count.
    load_set(0, 0);
    int k0 = 0;
    for (; k0 < K - 64; k0 += 64) {
        load_set(1, k0 + 32);
        compute_set(0);
        load_set(0, k0 + 64);
        compute_set(1);
    }
    load_set(1, K - 32);
    compute_set(0);
    compute_set(1);

    // Epilogue: activations + writes. C/D layout: n=lane%16, m=r+8*(lane/16).
    const int t_special = dir ? 0 : 2047;
    const int slot = layer * 2 + dir;
    #pragma unroll
    for (int mf = 0; mf < 4; ++mf) {
        #pragma unroll
        for (int nf = 0; nf < 2; ++nf) {
            const int n = j0 + nw + nf * 16 + ln15;
            #pragma unroll
            for (int r = 0; r < 8; ++r) {
                const int m = row0 + mw + mf * 16 + hl * 8 + r;
                float iv = acc[mf][nf][0][r];
                float gv = acc[mf][nf][1][r];
                float ov = acc[mf][nf][2][r];
                float c  = sigmoidf_(iv) * tanhf(gv);
                float h  = sigmoidf_(ov) * tanhf(c);
                size_t oidx = (size_t)m * 2048 + (size_t)dir * 1024 + n;
                if (OUT_BF16) ((uint16_t*)outp)[oidx] = f2bf(h);
                else          ((float*)outp)[oidx]    = h;
                if ((m >> 3) == t_special) {            // t = m / B, B = 8
                    const int b = m & 7;
                    hn[(size_t)slot * 8192 + b * 1024 + n] = h;
                    cn[(size_t)slot * 8192 + b * 1024 + n] = c;
                }
            }
        }
    }
}

extern "C" void kernel_launch(void* const* d_in, const int* in_sizes, int n_in,
                              void* d_out, int out_size, void* d_ws, size_t ws_size,
                              hipStream_t stream) {
    (void)in_sizes; (void)n_in; (void)out_size; (void)ws_size;

    const float* x   = (const float*)d_in[0];
    const float* Wf0 = (const float*)d_in[1];
    const float* bf0 = (const float*)d_in[2];
    const float* Wb0 = (const float*)d_in[3];
    const float* bb0 = (const float*)d_in[4];
    const float* Wf1 = (const float*)d_in[5];
    const float* bf1 = (const float*)d_in[6];
    const float* Wb1 = (const float*)d_in[7];
    const float* bb1 = (const float*)d_in[8];

    const int    M     = 16384;                 // S*B = 2048*8
    const size_t X_N   = (size_t)M * 1024;      // x elems
    const size_t W0_N  = (size_t)4096 * 1024;   // layer0 weight elems
    const size_t W1_N  = (size_t)4096 * 2048;   // layer1 weight elems
    const size_t OUT0_N= (size_t)M * 2048;      // inter-layer elems

    // Workspace layout (bf16 halves): ~151 MB total.
    uint16_t* ws      = (uint16_t*)d_ws;
    uint16_t* x_bf    = ws;  ws += X_N;
    uint16_t* wf0_bf  = ws;  ws += W0_N;
    uint16_t* wb0_bf  = ws;  ws += W0_N;
    uint16_t* wf1_bf  = ws;  ws += W1_N;
    uint16_t* wb1_bf  = ws;  ws += W1_N;
    uint16_t* out0_bf = ws;  ws += OUT0_N;

    cvt_f32_bf16_v4<<<2048, 256, 0, stream>>>((const float4*)x,   (ushort4*)x_bf,   (int)(X_N  / 4));
    cvt_f32_bf16_v4<<<1024, 256, 0, stream>>>((const float4*)Wf0, (ushort4*)wf0_bf, (int)(W0_N / 4));
    cvt_f32_bf16_v4<<<1024, 256, 0, stream>>>((const float4*)Wb0, (ushort4*)wb0_bf, (int)(W0_N / 4));
    cvt_f32_bf16_v4<<<2048, 256, 0, stream>>>((const float4*)Wf1, (ushort4*)wf1_bf, (int)(W1_N / 4));
    cvt_f32_bf16_v4<<<2048, 256, 0, stream>>>((const float4*)Wb1, (ushort4*)wb1_bf, (int)(W1_N / 4));

    float* out = (float*)d_out;                       // [2048, 8, 2048]
    float* hn  = out + (size_t)M * 2048;              // [4, 8, 1024]
    float* cn  = hn + (size_t)4 * 8 * 1024;           // [4, 8, 1024]

    dim3 grid(M / 256, 1024 / 64, 2), block(256);

    // Layer 0: K=1024, bf16 inter-layer output.
    lstm_layer_kernel<1024, true><<<grid, block, 0, stream>>>(
        x_bf, wf0_bf, wb0_bf, bf0, bb0, (void*)out0_bf, hn, cn, 0);

    // Layer 1: K=2048, fp32 final output.
    lstm_layer_kernel<2048, false><<<grid, block, 0, stream>>>(
        out0_bf, wf1_bf, wb1_bf, bf1, bb1, (void*)out, hn, cn, 1);
}
